// MoEFFN_56057913147552
// MI455X (gfx1250) — compile-verified, matrix operand order checked by reading, not run on hardware
//
#include <hip/hip_runtime.h>
#include <stdint.h>

// ---------------------------------------------------------------------------
// Types for CDNA5 WMMA (wave32, v_wmma_f32_16x16x32_bf16)
// ---------------------------------------------------------------------------
typedef __bf16   v16bf __attribute__((ext_vector_type(16)));
typedef __bf16   v2bf  __attribute__((ext_vector_type(2)));
typedef float    v8f   __attribute__((ext_vector_type(8)));
typedef float    f32x4 __attribute__((ext_vector_type(4)));
typedef float    f32x2 __attribute__((ext_vector_type(2)));
typedef unsigned int u32x4 __attribute__((ext_vector_type(4)));
typedef unsigned int u32x2 __attribute__((ext_vector_type(2)));

union FragU {
    v16bf v;
    struct { u32x4 lo; u32x4 hi; } u;
};

// Hardware f32->bf16 (RNE); backend may fuse pairs into v_cvt_pk_bf16_f32.
__device__ __forceinline__ unsigned short f2bf(float f) {
    __bf16 b = (__bf16)f;
    return __builtin_bit_cast(unsigned short, b);
}
// Two K-adjacent bf16 packed into one dword: lo -> [15:0], hi -> [31:16]
__device__ __forceinline__ unsigned int pack_bf16(float lo, float hi) {
    v2bf v;
    v[0] = (__bf16)lo;
    v[1] = (__bf16)hi;
    return __builtin_bit_cast(unsigned int, v);
}

__device__ __forceinline__ float fast_sigmoid(float x) {
    float e = __expf(-x);
#if __has_builtin(__builtin_amdgcn_rcpf)
    return __builtin_amdgcn_rcpf(1.0f + e);   // v_rcp_f32 (bf16 output anyway)
#else
    return 1.0f / (1.0f + e);
#endif
}

// Load one 16x32 (A) or 32x16-transposed (B) bf16 fragment from LDS.
// LDS layout: row-major [rows][32] bf16, 64B row stride.
// Lane<16  holds row (lane),    K = {0..7, 16..23}
// Lane>=16 holds row (lane-16), K = {8..15, 24..31}
__device__ __forceinline__ v16bf load_frag(const unsigned short* smem,
                                           int rowBase, int lane) {
    const int half = lane >> 4;
    const int rr   = lane & 15;
    const unsigned short* row = smem + (rowBase + rr) * 32 + half * 8;
    FragU f;
    f.u.lo = *(const u32x4*)(row);        // K k0 .. k0+7
    f.u.hi = *(const u32x4*)(row + 16);   // K k0+16 .. k0+23
    return f.v;
}

__device__ __forceinline__ v8f wmma_bf16(v16bf a, v16bf b, v8f c) {
    return __builtin_amdgcn_wmma_f32_16x16x32_bf16(
        false, a, false, b, (short)0, c, false, false);
}

// ---------------------------------------------------------------------------
// Constants
// ---------------------------------------------------------------------------
#define NTOK   8192       // B*T
#define DMODEL 2048
#define HCHUNK 1024       // routed H; shared H (8192) processed as 8 chunks
#define NEXP   8

// ---------------------------------------------------------------------------
// Staging helpers (256-thread workgroups)
// ---------------------------------------------------------------------------

// x[rowBase..+128, k0..+32] f32 -> bf16 LDS [128][32]; ds_store_b64 per thread.
__device__ __forceinline__ void stage_x_tile(const float* __restrict__ x,
                                             unsigned short* sA,
                                             int rowBase, int k0, int tid) {
#pragma unroll
    for (int p = 0; p < 4; ++p) {
        int idx = p * 256 + tid;          // 0..1023, 4 floats each
        int r   = idx >> 3;               // 0..127
        int kc  = (idx & 7) * 4;          // 0..28
        const float* src = x + (size_t)(rowBase + r) * DMODEL + k0 + kc;
        f32x4 v = *(const f32x4*)(src);
        if (k0 + 64 <= DMODEL - 32)
            __builtin_prefetch(src + 64, 0, 1);   // global_prefetch_b8, next+1 tile
        u32x2 d;
        d[0] = pack_bf16(v[0], v[1]);
        d[1] = pack_bf16(v[2], v[3]);
        *(u32x2*)(sA + r * 32 + kc) = d;
    }
}

// Two B matrices (gate & up), tile 32(K) x 64(N), transposed into LDS [64][32].
// 2x2 micro-transpose: packed dword stores (k, k+1).
__device__ __forceinline__ void stage_b2_tile(const float* __restrict__ Bg,
                                              const float* __restrict__ Bu,
                                              unsigned short* sBg,
                                              unsigned short* sBu,
                                              int ldb, int colBase, int k0, int tid) {
#pragma unroll
    for (int p = 0; p < 2; ++p) {
        int idx = p * 256 + tid;          // 0..511 over (16 kk) x (32 nn)
        int kk  = (idx >> 5) * 2;         // 0,2,..,30
        int nn  = (idx & 31) * 2;         // 0,2,..,62
        const size_t off = (size_t)(k0 + kk) * ldb + colBase + nn;
        f32x2 g0 = *(const f32x2*)(Bg + off);
        f32x2 g1 = *(const f32x2*)(Bg + off + ldb);
        f32x2 u0 = *(const f32x2*)(Bu + off);
        f32x2 u1 = *(const f32x2*)(Bu + off + ldb);
        *(unsigned int*)(sBg + (nn    ) * 32 + kk) = pack_bf16(g0[0], g1[0]);
        *(unsigned int*)(sBg + (nn + 1) * 32 + kk) = pack_bf16(g0[1], g1[1]);
        *(unsigned int*)(sBu + (nn    ) * 32 + kk) = pack_bf16(u0[0], u1[0]);
        *(unsigned int*)(sBu + (nn + 1) * 32 + kk) = pack_bf16(u0[1], u1[1]);
    }
}

// h (bf16) tile 128 x 32: raw 16B copies.
__device__ __forceinline__ void stage_h_tile(const unsigned short* __restrict__ hbuf,
                                             unsigned short* sA,
                                             int rowBase, int k0, int tid) {
#pragma unroll
    for (int p = 0; p < 2; ++p) {
        int idx = p * 256 + tid;          // 0..511, 8 u16 each
        int r   = idx >> 2;               // 0..127
        int kc  = (idx & 3) * 8;          // 0,8,16,24
        *(u32x4*)(sA + r * 32 + kc) =
            *(const u32x4*)(hbuf + (size_t)(rowBase + r) * HCHUNK + k0 + kc);
    }
}

// Bd tile 32(K) x 128(N), transposed into LDS [128][32] with 2x2 packing.
__device__ __forceinline__ void stage_bd_tile(const float* __restrict__ Bd,
                                              unsigned short* sB,
                                              int colBase, int k0, int tid) {
#pragma unroll
    for (int p = 0; p < 4; ++p) {
        int idx = p * 256 + tid;          // 0..1023 over (16 kk) x (64 nn)
        int kk  = (idx >> 6) * 2;         // 0,2,..,30
        int nn  = (idx & 63) * 2;         // 0,2,..,126
        const size_t off = (size_t)(k0 + kk) * DMODEL + colBase + nn;
        f32x2 b0 = *(const f32x2*)(Bd + off);
        f32x2 b1 = *(const f32x2*)(Bd + off + DMODEL);
        *(unsigned int*)(sB + (nn    ) * 32 + kk) = pack_bf16(b0[0], b1[0]);
        *(unsigned int*)(sB + (nn + 1) * 32 + kk) = pack_bf16(b0[1], b1[1]);
    }
}

// ---------------------------------------------------------------------------
// Gate: logits, softmax importance, top-2 combine weights.
// One wave (32 lanes) per token; 8 tokens per 256-thread workgroup.
// ---------------------------------------------------------------------------
__global__ __launch_bounds__(256)
void gate_kernel(const float* __restrict__ x, const float* __restrict__ wg,
                 float* __restrict__ combine, float* __restrict__ gImp) {
    __shared__ float sImp[NEXP];
    const int tid  = threadIdx.x;
    const int lane = tid & 31;
    const int wave = tid >> 5;
    if (tid < NEXP) sImp[tid] = 0.0f;
    __syncthreads();

    const int n = blockIdx.x * 8 + wave;
    float acc[NEXP];
#pragma unroll
    for (int e = 0; e < NEXP; ++e) acc[e] = 0.0f;

    for (int d = lane; d < DMODEL; d += 32) {
        float xv = x[(size_t)n * DMODEL + d];
#pragma unroll
        for (int e = 0; e < NEXP; ++e) acc[e] += xv * wg[e * DMODEL + d];
    }
#pragma unroll
    for (int e = 0; e < NEXP; ++e)
        for (int off = 16; off > 0; off >>= 1)
            acc[e] += __shfl_xor(acc[e], off, 32);

    if (lane == 0) {
        float m = acc[0];
#pragma unroll
        for (int e = 1; e < NEXP; ++e) m = fmaxf(m, acc[e]);
        float p[NEXP], s = 0.0f;
#pragma unroll
        for (int e = 0; e < NEXP; ++e) { p[e] = __expf(acc[e] - m); s += p[e]; }
        float inv = 1.0f / s;
#pragma unroll
        for (int e = 0; e < NEXP; ++e) atomicAdd(&sImp[e], p[e] * inv);
        // top-2
        int i1 = 0;
#pragma unroll
        for (int e = 1; e < NEXP; ++e) if (acc[e] > acc[i1]) i1 = e;
        int i2 = (i1 == 0) ? 1 : 0;
#pragma unroll
        for (int e = 0; e < NEXP; ++e)
            if (e != i1 && acc[e] > acc[i2]) i2 = e;
        float e2 = __expf(acc[i2] - acc[i1]);
        float g1 = 1.0f / (1.0f + e2);
        float g2 = e2 * g1;
#pragma unroll
        for (int e = 0; e < NEXP; ++e) combine[(size_t)n * NEXP + e] = 0.0f;
        combine[(size_t)n * NEXP + i1] = g1;
        combine[(size_t)n * NEXP + i2] = g2;
    }
    __syncthreads();
    if (tid < NEXP) atomicAdd(&gImp[tid], sImp[tid]);
}

// ---------------------------------------------------------------------------
// Gate-up GEMM: h = silu(x @ Bg) * (x @ Bu), bf16 out to hbuf [NTOK, 1024].
// Workgroup tile 128(M) x 64(N); 8 waves as 4(M) x 2(N); wave tile 32x32.
// Double-buffered LDS, one barrier per K-step.
// ---------------------------------------------------------------------------
__global__ __launch_bounds__(256)
void gateup_kernel(const float* __restrict__ x,
                   const float* __restrict__ Bg,
                   const float* __restrict__ Bu,
                   int ldb,
                   unsigned short* __restrict__ hbuf) {
    __shared__ unsigned short sA[2][128 * 32];
    __shared__ unsigned short sBg[2][64 * 32];
    __shared__ unsigned short sBu[2][64 * 32];

    const int tid  = threadIdx.x;
    const int lane = tid & 31;
    const int wave = tid >> 5;
    const int wm   = wave & 3;   // 0..3 -> M offset wm*32
    const int wn   = wave >> 2;  // 0..1 -> N offset wn*32
    const int rowBase = blockIdx.y * 128;
    const int colBase = blockIdx.x * 64;

    v8f accg[2][2] = {};
    v8f accu[2][2] = {};

    stage_x_tile(x, sA[0], rowBase, 0, tid);
    stage_b2_tile(Bg, Bu, sBg[0], sBu[0], ldb, colBase, 0, tid);
    __syncthreads();

    int cur = 0;
    for (int k0 = 0; k0 < DMODEL; k0 += 32, cur ^= 1) {
        v16bf af[2], bg[2], bu[2];
#pragma unroll
        for (int i = 0; i < 2; ++i) af[i] = load_frag(sA[cur], wm * 32 + i * 16, lane);
#pragma unroll
        for (int j = 0; j < 2; ++j) {
            bg[j] = load_frag(sBg[cur], wn * 32 + j * 16, lane);
            bu[j] = load_frag(sBu[cur], wn * 32 + j * 16, lane);
        }
        if (k0 + 32 < DMODEL) {   // stage next tile into the other buffer
            stage_x_tile(x, sA[cur ^ 1], rowBase, k0 + 32, tid);
            stage_b2_tile(Bg, Bu, sBg[cur ^ 1], sBu[cur ^ 1], ldb, colBase, k0 + 32, tid);
        }
#pragma unroll
        for (int i = 0; i < 2; ++i)
#pragma unroll
            for (int j = 0; j < 2; ++j) {
                accg[i][j] = wmma_bf16(af[i], bg[j], accg[i][j]);
                accu[i][j] = wmma_bf16(af[i], bu[j], accu[i][j]);
            }
        __syncthreads();
    }

    // ---- epilogue: silu(hg)*hu -> bf16 hbuf
    const int half = lane >> 4;
    const int cl   = lane & 15;
#pragma unroll
    for (int i = 0; i < 2; ++i)
#pragma unroll
        for (int j = 0; j < 2; ++j) {
            const int col = colBase + wn * 32 + j * 16 + cl;
#pragma unroll
            for (int r = 0; r < 8; ++r) {
                const int row = rowBase + wm * 32 + i * 16 + half * 8 + r;
                float g = accg[i][j][r];
                float u = accu[i][j][r];
                float h = g * fast_sigmoid(g) * u;   // silu(g)*u
                hbuf[(size_t)row * HCHUNK + col] = f2bf(h);
            }
        }
}

// ---------------------------------------------------------------------------
// Down GEMM: out[n,d] += scale[n] * sum_k h[n,k] * Bd[k,d], K = 1024.
// Workgroup tile 128(M) x 128(N); 8 waves as 4(M) x 2(N); wave tile 32x64.
// Double-buffered LDS. scale == nullptr -> 1.0 (shared expert).
// ---------------------------------------------------------------------------
__global__ __launch_bounds__(256)
void down_kernel(const unsigned short* __restrict__ hbuf,
                 const float* __restrict__ Bd,
                 float* __restrict__ out,
                 const float* __restrict__ scale,
                 int sstride) {
    __shared__ unsigned short sA[2][128 * 32];
    __shared__ unsigned short sB[2][128 * 32];

    const int tid  = threadIdx.x;
    const int lane = tid & 31;
    const int wave = tid >> 5;
    const int wm   = wave & 3;   // 0..3 -> M offset wm*32
    const int wn   = wave >> 2;  // 0..1 -> N offset wn*64
    const int rowBase = blockIdx.y * 128;
    const int colBase = blockIdx.x * 128;

    v8f acc[2][4] = {};

    stage_h_tile(hbuf, sA[0], rowBase, 0, tid);
    stage_bd_tile(Bd, sB[0], colBase, 0, tid);
    __syncthreads();

    int cur = 0;
    for (int k0 = 0; k0 < HCHUNK; k0 += 32, cur ^= 1) {
        v16bf af[2], bf[4];
#pragma unroll
        for (int i = 0; i < 2; ++i) af[i] = load_frag(sA[cur], wm * 32 + i * 16, lane);
#pragma unroll
        for (int j = 0; j < 4; ++j) bf[j] = load_frag(sB[cur], wn * 64 + j * 16, lane);
        if (k0 + 32 < HCHUNK) {
            stage_h_tile(hbuf, sA[cur ^ 1], rowBase, k0 + 32, tid);
            stage_bd_tile(Bd, sB[cur ^ 1], colBase, k0 + 32, tid);
        }
#pragma unroll
        for (int i = 0; i < 2; ++i)
#pragma unroll
            for (int j = 0; j < 4; ++j)
                acc[i][j] = wmma_bf16(af[i], bf[j], acc[i][j]);
        __syncthreads();
    }

    // ---- epilogue: out += scale * acc
    const int half = lane >> 4;
    const int cl   = lane & 15;
#pragma unroll
    for (int i = 0; i < 2; ++i)
#pragma unroll
        for (int r = 0; r < 8; ++r) {
            const int row = rowBase + wm * 32 + i * 16 + half * 8 + r;
            const float s = scale ? scale[(size_t)row * sstride] : 1.0f;
#pragma unroll
            for (int j = 0; j < 4; ++j) {
                const int col = colBase + wn * 64 + j * 16 + cl;
                const size_t o = (size_t)row * DMODEL + col;
                out[o] += s * acc[i][j][r];
            }
        }
}

// ---------------------------------------------------------------------------
// Load-balance loss: ce = importance / N * NEXP; loss = mean(ce^2)
// ---------------------------------------------------------------------------
__global__ void loss_kernel(const float* __restrict__ gImp,
                            float* __restrict__ out, size_t idx) {
    if (threadIdx.x == 0 && blockIdx.x == 0) {
        float s = 0.0f;
#pragma unroll
        for (int e = 0; e < NEXP; ++e) {
            float ce = gImp[e] * ((float)NEXP / (float)NTOK);
            s += ce * ce;
        }
        out[idx] = s / (float)NEXP;
    }
}

// ---------------------------------------------------------------------------
// Host launcher
// ---------------------------------------------------------------------------
extern "C" void kernel_launch(void* const* d_in, const int* in_sizes, int n_in,
                              void* d_out, int out_size, void* d_ws, size_t ws_size,
                              hipStream_t stream) {
    const float* x      = (const float*)d_in[0];   // [4,2048,2048]
    const float* w_gate = (const float*)d_in[1];   // [8,2048]
    const float* Wg     = (const float*)d_in[2];   // [8,2048,1024]
    const float* Wu     = (const float*)d_in[3];   // [8,2048,1024]
    const float* Wd     = (const float*)d_in[4];   // [8,1024,2048]
    const float* Sg     = (const float*)d_in[5];   // [2048,8192]
    const float* Su     = (const float*)d_in[6];   // [2048,8192]
    const float* Sd     = (const float*)d_in[7];   // [8192,2048]
    float* out = (float*)d_out;                    // [8192*2048 + 1]

    // workspace layout
    unsigned short* hbuf = (unsigned short*)d_ws;                 // 16 MiB bf16
    float* combine = (float*)((char*)d_ws + (size_t)NTOK * HCHUNK * 2);
    float* gImp    = combine + (size_t)NTOK * NEXP;

    hipMemsetAsync(d_out, 0, (size_t)out_size * sizeof(float), stream);
    hipMemsetAsync(gImp, 0, NEXP * sizeof(float), stream);

    gate_kernel<<<NTOK / 8, 256, 0, stream>>>(x, w_gate, combine, gImp);

    dim3 blk(256);
    dim3 ggu(HCHUNK / 64, NTOK / 128);   // (16, 64)
    dim3 gdn(DMODEL / 128, NTOK / 128);  // (16, 64)

    // shared expert: H_SHARED = 8192 as 8 chunks of 1024, combine weight = 1
    for (int c = 0; c < 8; ++c) {
        gateup_kernel<<<ggu, blk, 0, stream>>>(
            x, Sg + (size_t)c * HCHUNK, Su + (size_t)c * HCHUNK, 8192, hbuf);
        down_kernel<<<gdn, blk, 0, stream>>>(
            hbuf, Sd + (size_t)c * HCHUNK * DMODEL, out, nullptr, 0);
    }
    // routed experts (dense, zero-weight tokens contribute 0 — matches reference)
    for (int e = 0; e < NEXP; ++e) {
        gateup_kernel<<<ggu, blk, 0, stream>>>(
            x, Wg + (size_t)e * DMODEL * HCHUNK, Wu + (size_t)e * DMODEL * HCHUNK,
            HCHUNK, hbuf);
        down_kernel<<<gdn, blk, 0, stream>>>(
            hbuf, Wd + (size_t)e * HCHUNK * DMODEL, out, combine + e, NEXP);
    }

    loss_kernel<<<1, 1, 0, stream>>>(gImp, out, (size_t)NTOK * DMODEL);
}
